// BandMatrix_1726576853287
// MI455X (gfx1250) — matrix operand order repaired and checked by reading, not compiled, not run on hardware
//
#include <hip/hip_runtime.h>
#include <stdint.h>

#define IN_F   4096
#define OUT_F  4096
#define BWD    1024
#define BATCHN 8192

#define BM  128
#define BN  128
#define BK  32
#define LDPH 40   // bf16 elements per padded LDS row (32 data + 8 pad = 80 bytes)

typedef __attribute__((ext_vector_type(16))) __bf16          v16bf;
typedef __attribute__((ext_vector_type(8)))  float           v8f;
typedef __attribute__((ext_vector_type(4)))  float           f4;
typedef __attribute__((ext_vector_type(8)))  unsigned short  us8;

#if defined(__AMDGCN__) && __has_builtin(__builtin_amdgcn_global_load_async_to_lds_b128)
#define USE_ASYNC 1
#else
#define USE_ASYNC 0
#endif

#if USE_ASYNC
typedef __attribute__((__vector_size__(4 * sizeof(int)))) int v4i;
typedef __attribute__((address_space(1))) v4i as1_v4i;
typedef __attribute__((address_space(3))) v4i as3_v4i;
#endif

// ---- fp32 -> bf16 RNE (manual integer path; only used in preprocessing / fallback)
__device__ __forceinline__ unsigned short f2bf(float f) {
  unsigned u = __builtin_bit_cast(unsigned, f);
  u += 0x7FFFu + ((u >> 16) & 1u);
  return (unsigned short)(u >> 16);
}
__device__ __forceinline__ unsigned pk2(float a, float b) {
  return (unsigned)f2bf(a) | ((unsigned)f2bf(b) << 16);
}
union U8 { us8 v; unsigned u[4]; };
union Frag16 { v16bf v; us8 h[2]; };

__device__ __forceinline__ us8 pack8(const float* f) {
  U8 o;
#pragma unroll
  for (int j = 0; j < 4; ++j) o.u[j] = pk2(f[2 * j], f[2 * j + 1]);
  return o.v;
}

// ---- raw 16B copy global->LDS, async on gfx1250
__device__ __forceinline__ void async16(const unsigned short* g, unsigned short* l) {
#if USE_ASYNC
  __builtin_amdgcn_global_load_async_to_lds_b128(
      (as1_v4i*)(uintptr_t)g,
      (as3_v4i*)(unsigned)(uintptr_t)l, 0, 0);
#else
  *(us8*)l = *(const us8*)g;
#endif
}
__device__ __forceinline__ void wait_lds_copies() {
#if USE_ASYNC
#if __has_builtin(__builtin_amdgcn_s_wait_asynccnt)
  __builtin_amdgcn_s_wait_asynccnt(0);
#else
  asm volatile("s_wait_asynccnt 0" ::: "memory");
#endif
#endif
}

// ================= preprocessing: one-time fp32 -> bf16 (mask folded into W) ======
__global__ __launch_bounds__(256) void convert_x_bf16(
    const float* __restrict__ in, unsigned short* __restrict__ outp) {
  const int t = blockIdx.x * 256 + threadIdx.x;   // one 16-element chunk per thread
  const f4* p = (const f4*)(in + (size_t)t * 16);
  f4 a = p[0], b = p[1], c = p[2], d = p[3];
  float f[16] = {a.x,a.y,a.z,a.w, b.x,b.y,b.z,b.w, c.x,c.y,c.z,c.w, d.x,d.y,d.z,d.w};
  us8* o = (us8*)(outp + (size_t)t * 16);
  o[0] = pack8(f); o[1] = pack8(f + 8);
}

__global__ __launch_bounds__(256) void convert_w_bf16(
    const float* __restrict__ in, unsigned short* __restrict__ outp) {
  const int t = blockIdx.x * 256 + threadIdx.x;   // one 16-element chunk per thread
  const int row  = t >> 8;                        // 256 chunks per 4096-wide row
  const int colb = (t & 255) * 16;
  const f4* p = (const f4*)(in + (size_t)t * 16);
  f4 a = p[0], b = p[1], c = p[2], d = p[3];
  float f[16] = {a.x,a.y,a.z,a.w, b.x,b.y,b.z,b.w, c.x,c.y,c.z,c.w, d.x,d.y,d.z,d.w};
  const int lo = row - BWD, hi = row + BWD;
#pragma unroll
  for (int e = 0; e < 16; ++e) {
    const int col = colb + e;
    if (col < lo || col > hi) f[e] = 0.0f;        // band mask applied exactly once
  }
  us8* o = (us8*)(outp + (size_t)t * 16);
  o[0] = pack8(f); o[1] = pack8(f + 8);
}

// ================= banded GEMM, bf16 WMMA core =====================================
// PRE = true : sources are preconverted bf16 (xh/wh), staged via async-to-LDS.
// PRE = false: sources are fp32 (xf/wf), converted inline through a register pipeline.
template <bool PRE>
__global__ __launch_bounds__(256) void band_bf16_wmma(
    const float* __restrict__ xf, const float* __restrict__ wf,
    const unsigned short* __restrict__ xh, const unsigned short* __restrict__ wh,
    const float* __restrict__ bias, float* __restrict__ out) {
  __shared__ unsigned short ldsX[2][BM * LDPH];
  __shared__ unsigned short ldsW[2][BN * LDPH];

  const int tid  = threadIdx.x;
  const int wave = tid >> 5;
  const int lane = tid & 31;
  const int lr   = lane & 15;
  const int lh   = lane >> 4;
  const int wm   = wave & 1;       // 0..1 : 64-row (batch) slab
  const int wn   = wave >> 1;      // 0..3 : 32-col (out-feature) slab

  const int n0 = blockIdx.x * BN;
  const int m0 = blockIdx.y * BM;

  int kbeg = n0 - BWD;      if (kbeg < 0)    kbeg = 0;
  int kend = n0 + BN + BWD; if (kend > IN_F) kend = IN_F;
  const int nk = (kend - kbeg) / BK;

  v8f acc[4][2] = {};

  if constexpr (PRE) {
    // ---------- async bf16 staging: 512 chunks of 16B per tile, 2+2 per thread ----
    auto stage = [&](int b, int kb) {
#pragma unroll
      for (int i = 0; i < 2; ++i) {
        const int c = tid + i * 256;
        const int r = c >> 2, q = (c & 3) << 3;
        async16(xh + (size_t)(m0 + r) * IN_F + kb + q, &ldsX[b][r * LDPH + q]);
        async16(wh + (size_t)(n0 + r) * IN_F + kb + q, &ldsW[b][r * LDPH + q]);
      }
    };
    stage(0, kbeg);
    for (int ks = 0; ks < nk; ++ks) {
      const int buf = ks & 1;
      wait_lds_copies();
      __syncthreads();
      if (ks + 1 < nk) stage(buf ^ 1, kbeg + (ks + 1) * BK);

      Frag16 afr[4];
#pragma unroll
      for (int mt = 0; mt < 4; ++mt) {
        const unsigned short* rp = &ldsX[buf][(wm * 64 + mt * 16 + lr) * LDPH];
        const int k0 = lh ? 8 : 0;
        afr[mt].h[0] = *(const us8*)(rp + k0);
        afr[mt].h[1] = *(const us8*)(rp + k0 + 16);
      }
      Frag16 bfr[2];
#pragma unroll
      for (int nt = 0; nt < 2; ++nt) {
        const unsigned short* rp =
            &ldsW[buf][(wn * 32 + nt * 16 + lr) * LDPH + (lh ? 8 : 0)];
        bfr[nt].h[0] = *(const us8*)(rp);
        bfr[nt].h[1] = *(const us8*)(rp + 16);
      }
#pragma unroll
      for (int mt = 0; mt < 4; ++mt)
#pragma unroll
        for (int nt = 0; nt < 2; ++nt)
          acc[mt][nt] = __builtin_amdgcn_wmma_f32_16x16x32_bf16(
              false, afr[mt].v, false, bfr[nt].v, (short)0, acc[mt][nt],
              false, false);
    }
  } else {
    // ---------- fallback: fp32 sources, inline cvt through register pipeline ------
    const int sr = tid >> 1;
    const int sq = (tid & 1) * 16;
    const float* xrow = xf + (size_t)(m0 + sr) * IN_F;
    const float* wrow = wf + (size_t)(n0 + sr) * IN_F;
    const int nrow = n0 + sr;
    const int lo = nrow - BWD, hi = nrow + BWD;

    f4 rx[4], rw[4];
    {
      const f4* xp = (const f4*)(xrow + kbeg + sq);
      const f4* wp = (const f4*)(wrow + kbeg + sq);
#pragma unroll
      for (int j = 0; j < 4; ++j) { rx[j] = xp[j]; rw[j] = wp[j]; }
    }
    for (int ks = 0; ks < nk; ++ks) {
      const int buf = ks & 1;
      const int kb  = kbeg + ks * BK;
      {
        float fx[16], fw[16];
#pragma unroll
        for (int j = 0; j < 4; ++j) {
          fx[4*j] = rx[j].x; fx[4*j+1] = rx[j].y; fx[4*j+2] = rx[j].z; fx[4*j+3] = rx[j].w;
          fw[4*j] = rw[j].x; fw[4*j+1] = rw[j].y; fw[4*j+2] = rw[j].z; fw[4*j+3] = rw[j].w;
        }
#pragma unroll
        for (int e = 0; e < 16; ++e) {
          const int k = kb + sq + e;
          if (k < lo || k > hi) fw[e] = 0.0f;
        }
        unsigned short* xl = &ldsX[buf][sr * LDPH + sq];
        unsigned short* wl = &ldsW[buf][sr * LDPH + sq];
        *(us8*)(xl)     = pack8(fx);
        *(us8*)(xl + 8) = pack8(fx + 8);
        *(us8*)(wl)     = pack8(fw);
        *(us8*)(wl + 8) = pack8(fw + 8);
      }
      if (ks + 1 < nk) {
        const f4* xp = (const f4*)(xrow + kb + BK + sq);
        const f4* wp = (const f4*)(wrow + kb + BK + sq);
#pragma unroll
        for (int j = 0; j < 4; ++j) { rx[j] = xp[j]; rw[j] = wp[j]; }
      }
      __syncthreads();

      Frag16 afr[4];
#pragma unroll
      for (int mt = 0; mt < 4; ++mt) {
        const unsigned short* rp = &ldsX[buf][(wm * 64 + mt * 16 + lr) * LDPH];
        const int k0 = lh ? 8 : 0;
        afr[mt].h[0] = *(const us8*)(rp + k0);
        afr[mt].h[1] = *(const us8*)(rp + k0 + 16);
      }
      Frag16 bfr[2];
#pragma unroll
      for (int nt = 0; nt < 2; ++nt) {
        const unsigned short* rp =
            &ldsW[buf][(wn * 32 + nt * 16 + lr) * LDPH + (lh ? 8 : 0)];
        bfr[nt].h[0] = *(const us8*)(rp);
        bfr[nt].h[1] = *(const us8*)(rp + 16);
      }
#pragma unroll
      for (int mt = 0; mt < 4; ++mt)
#pragma unroll
        for (int nt = 0; nt < 2; ++nt)
          acc[mt][nt] = __builtin_amdgcn_wmma_f32_16x16x32_bf16(
              false, afr[mt].v, false, bfr[nt].v, (short)0, acc[mt][nt],
              false, false);
    }
  }

  // ---- Epilogue: VGPR v holds row (v + 8*lh), lane holds col lr; bias fused.
#pragma unroll
  for (int nt = 0; nt < 2; ++nt) {
    const int col = n0 + wn * 32 + nt * 16 + lr;
    const float b = bias[col];
#pragma unroll
    for (int mt = 0; mt < 4; ++mt) {
      const int rbase = m0 + wm * 64 + mt * 16 + lh * 8;
#pragma unroll
      for (int v = 0; v < 8; ++v)
        out[(size_t)(rbase + v) * OUT_F + col] = acc[mt][nt][v] + b;
    }
  }
}

extern "C" void kernel_launch(void* const* d_in, const int* in_sizes, int n_in,
                              void* d_out, int out_size, void* d_ws, size_t ws_size,
                              hipStream_t stream) {
  (void)in_sizes; (void)n_in; (void)out_size;
  const float* x    = (const float*)d_in[0];
  const float* wgt  = (const float*)d_in[1];
  const float* bias = (const float*)d_in[2];
  float* out = (float*)d_out;

  dim3 grid(OUT_F / BN, BATCHN / BM);   // N tiles fastest -> x-slab stays hot in L2

  const size_t needX = (size_t)BATCHN * IN_F * sizeof(unsigned short);
  const size_t needW = (size_t)OUT_F * IN_F * sizeof(unsigned short);
  if (ws_size >= needX + needW) {
    unsigned short* xh = (unsigned short*)d_ws;
    unsigned short* wh = xh + (size_t)BATCHN * IN_F;
    convert_x_bf16<<<(BATCHN * IN_F / 16) / 256, 256, 0, stream>>>(x, xh);
    convert_w_bf16<<<(OUT_F * IN_F / 16) / 256, 256, 0, stream>>>(wgt, wh);
    band_bf16_wmma<true><<<grid, 256, 0, stream>>>(nullptr, nullptr, xh, wh, bias, out);
  } else {
    band_bf16_wmma<false><<<grid, 256, 0, stream>>>(x, wgt, nullptr, nullptr, bias, out);
  }
}
